// FP8Linear_18717467476060
// MI455X (gfx1250) — 8-bit matrix operand order repaired and checked by reading, compiled, not run on hardware
//
#include <hip/hip_runtime.h>
#include <stdint.h>

typedef int   v16i __attribute__((ext_vector_type(16)));
typedef float v8f  __attribute__((ext_vector_type(8)));

#define FP8_MAX 448.0f
#define QBLK 128

__device__ __forceinline__ unsigned int f32_to_e4m3_sw(float f) {
  unsigned int u = __float_as_uint(f);
  unsigned int sgn = (u >> 24) & 0x80u;
  unsigned int a = u & 0x7fffffffu;
  if (a >= 0x7f800000u) return sgn | 0x7fu;
  float af = __uint_as_float(a);
  if (af >= 464.0f) return sgn | 0x7eu;
  int e = (int)(a >> 23) - 127;
  if (e < -6) {
    int m = (int)rintf(af * 512.0f);
    if (m >= 8) return sgn | 0x08u;
    return sgn | (unsigned int)m;
  }
  unsigned int mant = a & 0x7fffffu;
  unsigned int keep = mant >> 20;
  unsigned int rem  = mant & 0xfffffu;
  if (rem > 0x80000u || (rem == 0x80000u && (keep & 1u))) keep++;
  unsigned int ee = (unsigned int)(e + 7);
  if (keep == 8u) { keep = 0u; ee++; }
  if (ee > 15u || (ee == 15u && keep > 6u)) return sgn | 0x7eu;
  return sgn | (ee << 3) | keep;
}

__device__ __forceinline__ unsigned int pack4_e4m3(float x, float y, float z, float w) {
#if __has_builtin(__builtin_amdgcn_cvt_pk_fp8_f32)
  int lo = __builtin_amdgcn_cvt_pk_fp8_f32(x, y, 0, false);
  int hi = __builtin_amdgcn_cvt_pk_fp8_f32(z, w, lo, true);
  return (unsigned int)hi;
#else
  return  f32_to_e4m3_sw(x)        | (f32_to_e4m3_sw(y) << 8) |
         (f32_to_e4m3_sw(z) << 16) | (f32_to_e4m3_sw(w) << 24);
#endif
}

__global__ void wconv_kernel(const float* __restrict__ w, uint8_t* __restrict__ wq,
                             unsigned long long n4) {
  unsigned long long i = (unsigned long long)blockIdx.x * blockDim.x + threadIdx.x;
  if (i >= n4) return;
  float4 v = *reinterpret_cast<const float4*>(w + i * 4ull);
  reinterpret_cast<unsigned int*>(wq)[i] = pack4_e4m3(v.x, v.y, v.z, v.w);
}

__global__ void act_quant_kernel(const float* __restrict__ x,
                                 uint8_t* __restrict__ aq,
                                 float* __restrict__ as_t,
                                 int Kb, int M) {
  const int wave = (int)((blockIdx.x * blockDim.x + threadIdx.x) >> 5);
  const int lane = threadIdx.x & 31;
  const int m  = wave / Kb;
  const int kb = wave % Kb;
  const size_t K = (size_t)Kb * QBLK;

  const size_t src = (size_t)m * K + (size_t)kb * QBLK + (size_t)lane * 4;
  float4 v = *reinterpret_cast<const float4*>(x + src);
  float amax = fmaxf(fmaxf(fabsf(v.x), fabsf(v.y)), fmaxf(fabsf(v.z), fabsf(v.w)));
#pragma unroll
  for (int off = 16; off >= 1; off >>= 1)
    amax = fmaxf(amax, __shfl_xor(amax, off, 32));

  const float s   = amax * (1.0f / FP8_MAX);
  const float inv = (amax > 0.0f) ? (FP8_MAX / amax) : 0.0f;
  unsigned int packed = pack4_e4m3(v.x * inv, v.y * inv, v.z * inv, v.w * inv);

  const int h = (lane >> 2) & 1;
  const int vv = 4 * (lane >> 3) + (lane & 3);
  const int wl = (m & 15) + 16 * h;
  const size_t blk = ((size_t)(m >> 4) * Kb + kb) * 2048;
  *reinterpret_cast<unsigned int*>(aq + blk + (size_t)(wl * 16 + vv) * 4) = packed;

  if (lane == 0) as_t[(size_t)kb * M + m] = s;
}

__global__ void __launch_bounds__(256)
fp8_gemm_kernel(const uint8_t* __restrict__ aq, const float* __restrict__ as_t,
                const uint8_t* __restrict__ wq, const float* __restrict__ wscale,
                float* __restrict__ out, int M, int N, int K, int tiles_n) {
  constexpr int BROW = 144;
  __shared__ __align__(16) uint8_t bsh[2][64 * BROW];

  const int Kb   = K >> 7;
  const int tid  = threadIdx.x;
  const int wave = tid >> 5;
  const int lane = tid & 31;
  const int mr = lane & 15;
  const int h  = lane >> 4;

  const int wgm = blockIdx.x / tiles_n;
  const int wgn = blockIdx.x % tiles_n;
  const int m0  = wgm * 256 + wave * 32;
  const int n0  = wgn * 64;

  const uint8_t* ap0 = aq + ((size_t)(m0 >> 4) * Kb) * 2048 + (size_t)lane * 64;
  const uint8_t* ap1 = ap0 + (size_t)Kb * 2048;

  const float* wsrow = wscale + (size_t)(n0 >> 7) * Kb;
  const float* asp   = as_t + m0 + 8 * h;

  const int sr = tid >> 2;
  const int sc = (tid & 3) * 32;
  const uint8_t* bsrc = wq + (size_t)(n0 + sr) * K + sc;
  const int bdoff = sr * BROW + sc;

  {
    uint4 t0 = *reinterpret_cast<const uint4*>(bsrc);
    uint4 t1 = *reinterpret_cast<const uint4*>(bsrc + 16);
    *reinterpret_cast<uint4*>(&bsh[0][bdoff])      = t0;
    *reinterpret_cast<uint4*>(&bsh[0][bdoff + 16]) = t1;
  }
  __syncthreads();

  v8f acc[2][4];
#pragma unroll
  for (int i = 0; i < 2; ++i)
#pragma unroll
    for (int j = 0; j < 4; ++j)
#pragma unroll
      for (int v = 0; v < 8; ++v) acc[i][j][v] = 0.0f;

  for (int kb = 0; kb < Kb; ++kb) {
    const int  cur  = kb & 1;
    const bool more = (kb + 1) < Kb;

    uint4 t0, t1;
    if (more) {
      const uint8_t* p = bsrc + (size_t)(kb + 1) * 128;
      t0 = *reinterpret_cast<const uint4*>(p);
      t1 = *reinterpret_cast<const uint4*>(p + 16);
    }

    const size_t ao = (size_t)kb * 2048;
    __builtin_prefetch(ap0 + ao + 2048, 0, 0);

    v16i a0, a1;
#pragma unroll
    for (int g = 0; g < 4; ++g) {
      uint4 d;
      d = *reinterpret_cast<const uint4*>(ap0 + ao + g * 16);
      a0[4*g] = (int)d.x; a0[4*g+1] = (int)d.y; a0[4*g+2] = (int)d.z; a0[4*g+3] = (int)d.w;
      d = *reinterpret_cast<const uint4*>(ap1 + ao + g * 16);
      a1[4*g] = (int)d.x; a1[4*g+1] = (int)d.y; a1[4*g+2] = (int)d.z; a1[4*g+3] = (int)d.w;
    }

    v16i b0, b1, b2, b3;
    const uint8_t* bb = &bsh[cur][mr * BROW + h * 16];
#pragma unroll
    for (int g = 0; g < 4; ++g) {
      uint4 d;
      d = *reinterpret_cast<const uint4*>(bb + g * 32);
      b0[4*g] = (int)d.x; b0[4*g+1] = (int)d.y; b0[4*g+2] = (int)d.z; b0[4*g+3] = (int)d.w;
      d = *reinterpret_cast<const uint4*>(bb + 16 * BROW + g * 32);
      b1[4*g] = (int)d.x; b1[4*g+1] = (int)d.y; b1[4*g+2] = (int)d.z; b1[4*g+3] = (int)d.w;
      d = *reinterpret_cast<const uint4*>(bb + 32 * BROW + g * 32);
      b2[4*g] = (int)d.x; b2[4*g+1] = (int)d.y; b2[4*g+2] = (int)d.z; b2[4*g+3] = (int)d.w;
      d = *reinterpret_cast<const uint4*>(bb + 48 * BROW + g * 32);
      b3[4*g] = (int)d.x; b3[4*g+1] = (int)d.y; b3[4*g+2] = (int)d.z; b3[4*g+3] = (int)d.w;
    }

    const float ws = wsrow[kb];
    float4 c00 = *reinterpret_cast<const float4*>(asp);
    float4 c01 = *reinterpret_cast<const float4*>(asp + 4);
    float4 c10 = *reinterpret_cast<const float4*>(asp + 16);
    float4 c11 = *reinterpret_cast<const float4*>(asp + 20);
    asp += M;
    float cs0[8] = { c00.x * ws, c00.y * ws, c00.z * ws, c00.w * ws,
                     c01.x * ws, c01.y * ws, c01.z * ws, c01.w * ws };
    float cs1[8] = { c10.x * ws, c10.y * ws, c10.z * ws, c10.w * ws,
                     c11.x * ws, c11.y * ws, c11.z * ws, c11.w * ws };

    v8f z = {};
    v8f p;
    p = __builtin_amdgcn_wmma_f32_16x16x128_fp8_fp8(a0, b0, (short)0, z, false, false);
#pragma unroll
    for (int v = 0; v < 8; ++v) acc[0][0][v] += p[v] * cs0[v];
    p = __builtin_amdgcn_wmma_f32_16x16x128_fp8_fp8(a0, b1, (short)0, z, false, false);
#pragma unroll
    for (int v = 0; v < 8; ++v) acc[0][1][v] += p[v] * cs0[v];
    p = __builtin_amdgcn_wmma_f32_16x16x128_fp8_fp8(a0, b2, (short)0, z, false, false);
#pragma unroll
    for (int v = 0; v < 8; ++v) acc[0][2][v] += p[v] * cs0[v];
    p = __builtin_amdgcn_wmma_f32_16x16x128_fp8_fp8(a0, b3, (short)0, z, false, false);
#pragma unroll
    for (int v = 0; v < 8; ++v) acc[0][3][v] += p[v] * cs0[v];
    p = __builtin_amdgcn_wmma_f32_16x16x128_fp8_fp8(a1, b0, (short)0, z, false, false);
#pragma unroll
    for (int v = 0; v < 8; ++v) acc[1][0][v] += p[v] * cs1[v];
    p = __builtin_amdgcn_wmma_f32_16x16x128_fp8_fp8(a1, b1, (short)0, z, false, false);
#pragma unroll
    for (int v = 0; v < 8; ++v) acc[1][1][v] += p[v] * cs1[v];
    p = __builtin_amdgcn_wmma_f32_16x16x128_fp8_fp8(a1, b2, (short)0, z, false, false);
#pragma unroll
    for (int v = 0; v < 8; ++v) acc[1][2][v] += p[v] * cs1[v];
    p = __builtin_amdgcn_wmma_f32_16x16x128_fp8_fp8(a1, b3, (short)0, z, false, false);
#pragma unroll
    for (int v = 0; v < 8; ++v) acc[1][3][v] += p[v] * cs1[v];

    if (more) {
      *reinterpret_cast<uint4*>(&bsh[cur ^ 1][bdoff])      = t0;
      *reinterpret_cast<uint4*>(&bsh[cur ^ 1][bdoff + 16]) = t1;
    }
    __syncthreads();
  }

#pragma unroll
  for (int dm = 0; dm < 2; ++dm)
#pragma unroll
    for (int dn = 0; dn < 4; ++dn)
#pragma unroll
      for (int v = 0; v < 8; ++v)
        out[(size_t)(m0 + dm * 16 + 8 * h + v) * N + (n0 + dn * 16 + mr)] = acc[dm][dn][v];
}

extern "C" void kernel_launch(void* const* d_in, const int* in_sizes, int n_in,
                              void* d_out, int out_size, void* d_ws, size_t ws_size,
                              hipStream_t stream) {
  const float* x      = (const float*)d_in[0];
  const float* w      = (const float*)d_in[1];
  const float* wscale = (const float*)d_in[2];

  const int K  = 4096;
  const int M  = in_sizes[0] / K;
  const int N  = in_sizes[1] / K;
  const int Kb = K / QBLK;

  uint8_t* wq = (uint8_t*)d_ws;
  uint8_t* aq = wq + (size_t)N * K;
  float*   as = (float*)(aq + (size_t)M * K);

  {
    unsigned long long n4 = (unsigned long long)N * K / 4;
    int blk = (int)((n4 + 255ull) / 256ull);
    wconv_kernel<<<blk, 256, 0, stream>>>(w, wq, n4);
  }
  {
    int waves = M * Kb;
    int blk = waves * 32 / 256;
    act_quant_kernel<<<blk, 256, 0, stream>>>(x, aq, as, Kb, M);
  }
  {
    int tiles_n = N / 64;
    int blk     = (M / 256) * tiles_n;
    fp8_gemm_kernel<<<blk, 256, 0, stream>>>(aq, as, wq, wscale,
                                             (float*)d_out, M, N, K, tiles_n);
  }
}
